// MultiheadDiffAttn_42021960024164
// MI455X (gfx1250) — matrix operand order repaired and checked by reading, not compiled, not run on hardware
//
#include <hip/hip_runtime.h>

// ---------------------------------------------------------------------------
// MI455X (gfx1250) DiffAttn forward. wave32, WMMA bf16 16x16x32, f32 accum.
// NT (non-temporal) hints on the 134MB diff stream + one-shot f32 traffic so
// the reused bf16 working set (~50MB) stays resident in the 192MB L2.
// ---------------------------------------------------------------------------

typedef __attribute__((ext_vector_type(16))) __bf16 bf16x16;
typedef __attribute__((ext_vector_type(8)))  __bf16 bf16x8;
typedef __attribute__((ext_vector_type(4)))  __bf16 bf16x4;
typedef __attribute__((ext_vector_type(8)))  float  f32x8;
typedef __attribute__((ext_vector_type(4)))  float  f32x4;

#define LAMBDA_INIT 0.78360576653f
#define POST_SCALE  0.21639423347f   // 1 - LAMBDA_INIT

// A-fragment (16x32 bf16, row-major source, K contiguous).
// ISA layout: lane holds row M=lane%16; K = (e<8?0:16) + (lane/16)*8 + e%8.
__device__ __forceinline__ bf16x16 frag_a(const __bf16* p0, int lda, int lane) {
  const __bf16* p = p0 + (size_t)(lane & 15) * lda + ((lane >> 4) << 3);
  bf16x8 lo = *(const bf16x8*)(p);        // K = kb .. kb+7
  bf16x8 hi = *(const bf16x8*)(p + 16);   // K = kb+16 .. kb+23
  bf16x16 r;
#pragma unroll
  for (int i = 0; i < 8; ++i) { r[i] = lo[i]; r[i + 8] = hi[i]; }
  return r;
}

// B-fragment (32x16 bf16) from B^T stored row-major [N][K] (K contiguous).
// ISA layout: lane holds column N=lane%16; K = (e<8?0:16) + (lane/16)*8 + e%8.
__device__ __forceinline__ bf16x16 frag_b(const __bf16* p0, int ldb, int lane) {
  const __bf16* p = p0 + (size_t)(lane & 15) * ldb + ((lane >> 4) << 3);
  bf16x8 lo = *(const bf16x8*)(p);
  bf16x8 hi = *(const bf16x8*)(p + 16);
  bf16x16 r;
#pragma unroll
  for (int i = 0; i < 8; ++i) { r[i] = lo[i]; r[i + 8] = hi[i]; }
  return r;
}

__device__ __forceinline__ f32x8 wmma_bf16(bf16x16 a, bf16x16 b, f32x8 c) {
  return __builtin_amdgcn_wmma_f32_16x16x32_bf16(false, a, false, b,
                                                 (short)0, c, false, false);
}

// ---------------------------------------------------------------------------
// Elementwise f32 -> bf16 (x input). NT load: f32 source is read exactly once.
// ---------------------------------------------------------------------------
__global__ __launch_bounds__(256) void convert_f32_to_bf16(
    const float* __restrict__ src, __bf16* __restrict__ dst, int n) {
  int i = (blockIdx.x * 256 + threadIdx.x) * 4;
  if (i + 3 < n) {
    f32x4 v = __builtin_nontemporal_load((const f32x4*)(src + i));
    bf16x4 o;
#pragma unroll
    for (int j = 0; j < 4; ++j) o[j] = (__bf16)v[j];
    *(bf16x4*)(dst + i) = o;   // bf16 activations stay RT (reused by 3 GEMMs)
  }
}

// ---------------------------------------------------------------------------
// Transposing f32 -> bf16 converter: Wt[n][k] = W[k][n] * scale.
// LDS 32x33 tile, block (32,8). NT loads on the one-shot f32 weights.
// ---------------------------------------------------------------------------
__global__ __launch_bounds__(256) void transpose_convert_bf16(
    const float* __restrict__ W, __bf16* __restrict__ Wt, int K, int N,
    float scale) {
  __shared__ float tile[32][33];
  const int n0 = blockIdx.x << 5;
  const int k0 = blockIdx.y << 5;
#pragma unroll
  for (int i = threadIdx.y; i < 32; i += 8)
    tile[i][threadIdx.x] =
        __builtin_nontemporal_load(&W[(size_t)(k0 + i) * N + n0 + threadIdx.x]);
  __syncthreads();
#pragma unroll
  for (int i = threadIdx.y; i < 32; i += 8)
    Wt[(size_t)(n0 + i) * K + k0 + threadIdx.x] =
        (__bf16)(tile[threadIdx.x][i] * scale);
}

// ---------------------------------------------------------------------------
// Generic WMMA GEMM: C[M,N] = A[M,K](bf16,row-major) x Bt[N,K](bf16,row-major)
// Block = 256 threads (8 waves), 128x128 tile, each wave 32x64 (2x4 WMMA).
// MODE 0: bf16 out, row-major ldc
// MODE 1: bf16 out, V-transpose layout [b,kv,j,t]  (M=b*1024+t, N=kv*128+j)
// MODE 2: f32 out, row-major ldc, non-temporal (final output, never re-read)
// ---------------------------------------------------------------------------
template <int MODE>
__global__ __launch_bounds__(256) void gemm_bf16_wmma(
    const __bf16* __restrict__ A, const __bf16* __restrict__ Bt,
    void* __restrict__ Cv, int K, int lda, int ldb, int ldc) {
  const int lane = threadIdx.x & 31;
  const int w = threadIdx.x >> 5;
  const int m0 = blockIdx.y * 128 + (w & 3) * 32;
  const int n0 = blockIdx.x * 128 + (w >> 2) * 64;
  const __bf16* Abase = A + (size_t)m0 * lda;
  const __bf16* Bbase = Bt + (size_t)n0 * ldb;
  f32x8 acc[2][4] = {};
  for (int kk = 0; kk < K; kk += 32) {
    if (kk + 256 < K) {   // prefetch ahead on K -> global_prefetch_b8
      __builtin_prefetch(Abase + (size_t)(lane & 15) * lda + kk + 256, 0, 3);
      __builtin_prefetch(Bbase + (size_t)(lane & 15) * ldb + kk + 256, 0, 3);
    }
    bf16x16 af[2];
#pragma unroll
    for (int mt = 0; mt < 2; ++mt)
      af[mt] = frag_a(Abase + (size_t)(mt * 16) * lda + kk, lda, lane);
#pragma unroll
    for (int nt = 0; nt < 4; ++nt) {
      bf16x16 bfr = frag_b(Bbase + (size_t)(nt * 16) * ldb + kk, ldb, lane);
#pragma unroll
      for (int mt = 0; mt < 2; ++mt)
        acc[mt][nt] = wmma_bf16(af[mt], bfr, acc[mt][nt]);
    }
  }
  // C layout: element r of v8f -> M = r + 8*(lane/16), N = lane%16
  const int lm = lane & 15, lh = lane >> 4;
#pragma unroll
  for (int mt = 0; mt < 2; ++mt)
#pragma unroll
    for (int nt = 0; nt < 4; ++nt)
#pragma unroll
      for (int r = 0; r < 8; ++r) {
        const int m = m0 + mt * 16 + (lh << 3) + r;
        const int n = n0 + nt * 16 + lm;
        const float v = acc[mt][nt][r];
        if (MODE == 0) {
          ((__bf16*)Cv)[(size_t)m * ldc + n] = (__bf16)v;
        } else if (MODE == 1) {
          const int bb = m >> 10, t = m & 1023;
          const int kvh = n >> 7, j = n & 127;
          ((__bf16*)Cv)[((((size_t)(bb * 8 + kvh) << 7) + j) << 10) + t] =
              (__bf16)v;
        } else {
          __builtin_nontemporal_store(v, &((float*)Cv)[(size_t)m * ldc + n]);
        }
      }
}

// ---------------------------------------------------------------------------
// Fused differential attention. Grid (T/16, H=16, B=2), block 128 (4 waves).
// Dynamic LDS: s1[16x1024]f32 | s2[16x1024]f32 | db[16x1024]bf16 | rowinv[32]
// ---------------------------------------------------------------------------
__global__ __launch_bounds__(128) void diff_attn_fused(
    const __bf16* __restrict__ qb, const __bf16* __restrict__ kb,
    const __bf16* __restrict__ vt, const float* __restrict__ lq1,
    const float* __restrict__ lk1, const float* __restrict__ lq2,
    const float* __restrict__ lk2, const float* __restrict__ subw,
    __bf16* __restrict__ attnb, float* __restrict__ diffOut) {
  extern __shared__ char smem[];
  float* s1 = (float*)smem;                 // sub-head 0 scores/probs
  float* s2 = (float*)(smem + 65536);       // sub-head 1 scores/probs
  __bf16* db = (__bf16*)(smem + 131072);    // bf16 diff tile
  float* rowinv = (float*)(smem + 163840);  // 32 x 1/rowsum

  const int tid = threadIdx.x;
  const int lane = tid & 31;
  const int w = tid >> 5;
  const int R0 = blockIdx.x << 4;
  const int h = blockIdx.y;
  const int b = blockIdx.z;
  const int nct = blockIdx.x + 1;       // causal column tiles
  const int cl = nct << 4;              // covered columns (multiple of 16)
  const int kupc = (cl + 31) & ~31;
  const int kup = kupc < 1024 ? kupc : 1024;
  const int kv = h >> 1;

  // lambda reparameterization (64-element dots; trivial, done per thread)
  float l1 = 0.f, l2 = 0.f;
#pragma unroll
  for (int i = 0; i < 64; ++i) { l1 += lq1[i] * lk1[i]; l2 += lq2[i] * lk2[i]; }
  const float lam = __expf(l1) - __expf(l2) + LAMBDA_INIT;

  // ---- scores via WMMA for both sub-heads (q pre-scaled by 1/8) ----
  const __bf16* qbase = qb + ((size_t)(b * 1024 + R0) << 11);
  bf16x16 afr[2][2];
#pragma unroll
  for (int sh = 0; sh < 2; ++sh)
#pragma unroll
    for (int kx = 0; kx < 2; ++kx)
      afr[sh][kx] = frag_a(qbase + (2 * h + sh) * 64 + kx * 32, 2048, lane);

  const int lm = lane & 15, lh = lane >> 4;
  for (int ct = w; ct < nct; ct += 4) {
    const int c0 = ct << 4;
    const __bf16* kbase = kb + ((size_t)(b * 1024 + c0) << 10) + h * 64;
#pragma unroll
    for (int sh = 0; sh < 2; ++sh) {
      f32x8 acc = {};
#pragma unroll
      for (int kx = 0; kx < 2; ++kx) {
        bf16x16 bfr = frag_b(kbase + kx * 32, 1024, lane);
        acc = wmma_bf16(afr[sh][kx], bfr, acc);
      }
      float* sb = sh ? s2 : s1;
#pragma unroll
      for (int r = 0; r < 8; ++r)
        sb[((r + (lh << 3)) << 10) + c0 + lm] = acc[r];
    }
  }
  __syncthreads();

  // ---- softmax in LDS: 32 (row,buf) units, 4 lanes each, f32x4 chunks ----
  {
    const int u = tid >> 2;
    const int part = tid & 3;
    const int r = u & 15;
    float* sb = (u < 16) ? s1 : s2;
    float* row = sb + (r << 10);
    const int tg = R0 + r;  // last valid (causal) column

    // mask the single boundary tile [R0, R0+16): each lane owns 4 columns
    {
      const int s0 = R0 + part * 4;
#pragma unroll
      for (int j = 0; j < 4; ++j)
        if (s0 + j > tg) row[s0 + j] = -3.0e38f;
    }
    // guard-free vectorized max over covered region [0, cl)
    float mx = -3.0e38f;
    for (int s4 = part * 4; s4 < cl; s4 += 16) {
      f32x4 v = *(const f32x4*)(row + s4);
      mx = fmaxf(mx, fmaxf(fmaxf(v[0], v[1]), fmaxf(v[2], v[3])));
    }
    mx = fmaxf(mx, __shfl_xor(mx, 1));
    mx = fmaxf(mx, __shfl_xor(mx, 2));
    // exp + sum (masked entries: exp(-3e38 - mx) == 0)
    float sm = 0.f;
    for (int s4 = part * 4; s4 < cl; s4 += 16) {
      f32x4 v = *(const f32x4*)(row + s4);
      f32x4 p;
#pragma unroll
      for (int j = 0; j < 4; ++j) p[j] = __expf(v[j] - mx);
      *(f32x4*)(row + s4) = p;
      sm += (p[0] + p[1]) + (p[2] + p[3]);
    }
    sm += __shfl_xor(sm, 1);
    sm += __shfl_xor(sm, 2);
    if (part == 0) rowinv[u] = 1.0f / sm;
  }
  __syncthreads();

  // ---- diff = p1 - lam*p2 : NT f32x4 to d_out, bf16x4 tile to LDS ----
  {
    const int r = tid >> 3;
    const int p8 = tid & 7;
    const int tg = R0 + r;
    const float i1 = rowinv[r], i2 = rowinv[16 + r];
    float* drow = diffOut + ((((size_t)(b * 16 + h) << 10) + tg) << 10);
    for (int s4 = p8 * 4; s4 < 1024; s4 += 32) {
      f32x4 d = {0.f, 0.f, 0.f, 0.f};
      if (s4 < cl) {  // probs already 0 in masked boundary columns
        f32x4 a = *(const f32x4*)(s1 + (r << 10) + s4);
        f32x4 c = *(const f32x4*)(s2 + (r << 10) + s4);
#pragma unroll
        for (int j = 0; j < 4; ++j) d[j] = a[j] * i1 - lam * (c[j] * i2);
      }
      __builtin_nontemporal_store(d, (f32x4*)(drow + s4));
      if (s4 < kup) {  // also zero-fills db on [cl, kup)
        bf16x4 o;
#pragma unroll
        for (int j = 0; j < 4; ++j) o[j] = (__bf16)d[j];
        *(bf16x4*)(db + (r << 10) + s4) = o;
      }
    }
  }
  __syncthreads();

  // ---- attn = diff @ V : 16x128 via WMMA, V pre-transposed [b,kv,j,t] ----
  {
    f32x8 acc[2] = {};
    const __bf16* vbase = vt + (((size_t)(b * 8 + kv) << 7) << 10);
    for (int kt = 0; kt < kup; kt += 32) {
      bf16x16 af = frag_a(db + kt, 1024, lane);
#pragma unroll
      for (int i = 0; i < 2; ++i) {
        const int nt = w + (i << 2);
        bf16x16 bfr = frag_b(vbase + ((size_t)(nt << 4) << 10) + kt, 1024, lane);
        acc[i] = wmma_bf16(af, bfr, acc[i]);
      }
    }
    float* attnf = s2;  // reuse: 16 x 128 f32
#pragma unroll
    for (int i = 0; i < 2; ++i) {
      const int nt = w + (i << 2);
#pragma unroll
      for (int r = 0; r < 8; ++r)
        attnf[((r + (lh << 3)) << 7) + (nt << 4) + lm] = acc[i][r];
    }
  }
  __syncthreads();

  // ---- RMSNorm(128) * subln_w * (1 - lambda_init), store bf16 ----
  {
    float* attnf = s2;
    const int r = tid >> 3;
    const int p8 = tid & 7;
    const int j0 = p8 << 4;  // each lane owns 16 contiguous columns
    f32x4 v[4];
    float ss = 0.f;
#pragma unroll
    for (int c = 0; c < 4; ++c) {
      v[c] = *(const f32x4*)(attnf + (r << 7) + j0 + c * 4);
#pragma unroll
      for (int j = 0; j < 4; ++j) ss += v[c][j] * v[c][j];
    }
    ss += __shfl_xor(ss, 1);
    ss += __shfl_xor(ss, 2);
    ss += __shfl_xor(ss, 4);
    const float rms = rsqrtf(ss * 0.0078125f + 1e-5f);
    __bf16* orow = attnb + ((size_t)(b * 1024 + R0 + r) << 11) + (h << 7) + j0;
    bf16x8 o[2];
#pragma unroll
    for (int c = 0; c < 4; ++c)
#pragma unroll
      for (int j = 0; j < 4; ++j)
        o[c >> 1][((c & 1) << 2) + j] =
            (__bf16)(v[c][j] * rms * subw[j0 + c * 4 + j] * POST_SCALE);
    *(bf16x8*)(orow) = o[0];
    *(bf16x8*)(orow + 8) = o[1];
  }
}

// ---------------------------------------------------------------------------
// Host orchestration
// ---------------------------------------------------------------------------
extern "C" void kernel_launch(void* const* d_in, const int* in_sizes, int n_in,
                              void* d_out, int out_size, void* d_ws,
                              size_t ws_size, hipStream_t stream) {
  (void)in_sizes; (void)n_in; (void)out_size; (void)ws_size;
  const float* x   = (const float*)d_in[0];
  const float* Wq  = (const float*)d_in[1];
  const float* Wk  = (const float*)d_in[2];
  const float* Wv  = (const float*)d_in[3];
  const float* Wo  = (const float*)d_in[4];
  const float* lq1 = (const float*)d_in[5];
  const float* lk1 = (const float*)d_in[6];
  const float* lq2 = (const float*)d_in[7];
  const float* lk2 = (const float*)d_in[8];
  const float* sw  = (const float*)d_in[9];

  __bf16* xb    = (__bf16*)d_ws;                        // [2048,2048]
  __bf16* wqT   = xb    + (size_t)4 * 1024 * 1024;      // [2048,2048] (N,K)
  __bf16* wkT   = wqT   + (size_t)4 * 1024 * 1024;      // [1024,2048]
  __bf16* wvT   = wkT   + (size_t)2 * 1024 * 1024;      // [1024,2048]
  __bf16* woT   = wvT   + (size_t)2 * 1024 * 1024;      // [2048,2048]
  __bf16* qbuf  = woT   + (size_t)4 * 1024 * 1024;      // [2048,2048]
  __bf16* kbuf  = qbuf  + (size_t)4 * 1024 * 1024;      // [2048,1024]
  __bf16* vtb   = kbuf  + (size_t)2 * 1024 * 1024;      // [2,8,128,1024]
  __bf16* attnb = vtb   + (size_t)2 * 1024 * 1024;      // [2048,2048]

  float* out = (float*)d_out;
  float* diffOut = out + (size_t)2 * 1024 * 2048;

  // 1. x -> bf16
  convert_f32_to_bf16<<<dim3(4096), dim3(256), 0, stream>>>(
      x, xb, 4 * 1024 * 1024);
  // 2. weights -> bf16 transposed Bt[n][k]; fold q scaling (1/sqrt(64)) in Wq
  transpose_convert_bf16<<<dim3(64, 64), dim3(32, 8), 0, stream>>>(
      Wq, wqT, 2048, 2048, 0.125f);
  transpose_convert_bf16<<<dim3(32, 64), dim3(32, 8), 0, stream>>>(
      Wk, wkT, 2048, 1024, 1.0f);
  transpose_convert_bf16<<<dim3(32, 64), dim3(32, 8), 0, stream>>>(
      Wv, wvT, 2048, 1024, 1.0f);
  transpose_convert_bf16<<<dim3(64, 64), dim3(32, 8), 0, stream>>>(
      Wo, woT, 2048, 2048, 1.0f);
  // 3. projections (WMMA): q [2048,2048], k [2048,1024], v -> transposed
  gemm_bf16_wmma<0><<<dim3(16, 16), dim3(256), 0, stream>>>(
      xb, wqT, qbuf, 2048, 2048, 2048, 2048);
  gemm_bf16_wmma<0><<<dim3(8, 16), dim3(256), 0, stream>>>(
      xb, wkT, kbuf, 2048, 2048, 2048, 1024);
  gemm_bf16_wmma<1><<<dim3(8, 16), dim3(256), 0, stream>>>(
      xb, wvT, vtb, 2048, 2048, 2048, 0);
  // 4. fused dual-softmax differential attention (+diff output)
  diff_attn_fused<<<dim3(64, 16, 2), dim3(128), 163968, stream>>>(
      qbuf, kbuf, vtb, lq1, lk1, lq2, lk2, sw, attnb, diffOut);
  // 5. output projection -> f32 d_out (non-temporal)
  gemm_bf16_wmma<2><<<dim3(16, 16), dim3(256), 0, stream>>>(
      attnb, woT, out, 2048, 2048, 2048, 2048);
}
